// GroupedQueryAttention_85315230368397
// MI455X (gfx1250) — matrix operand order repaired and checked by reading, not compiled, not run on hardware
//
#include <hip/hip_runtime.h>
#include <hip/hip_bf16.h>

// ---------------------------------------------------------------------------
// GQA forward for MI455X (gfx1250): bf16 WMMA everywhere, f32 accumulate.
// B=1, S=2048, HID=4096, NH=32, NKV=8, HD=128, N_REP=4, causal, RoPE.
// ---------------------------------------------------------------------------

#define S_LEN 2048
#define HIDN  4096
#define NH    32
#define NKV   8
#define HD    128

typedef __attribute__((ext_vector_type(16))) __bf16 v16bf;
typedef __attribute__((ext_vector_type(8)))  float  v8f;
typedef __attribute__((ext_vector_type(4)))  int    i32x4;

// --------------------------- async LDS copy path ---------------------------
#if defined(__has_builtin)
#if __has_builtin(__builtin_amdgcn_global_load_async_to_lds_b128) && \
    __has_builtin(__builtin_amdgcn_s_wait_asynccnt)
#define USE_ASYNC_LDS 1
#endif
#endif

#if defined(USE_ASYNC_LDS)
typedef __attribute__((address_space(1))) i32x4 ga_i32x4;   // global
typedef __attribute__((address_space(3))) i32x4 ls_i32x4;   // LDS
// 16B global -> LDS copy, tracked with ASYNCcnt (no VGPR bounce).
__device__ __forceinline__ void async_copy16(const unsigned short* gsrc,
                                             unsigned short* ldst) {
  __builtin_amdgcn_global_load_async_to_lds_b128(
      (ga_i32x4*)gsrc, (ls_i32x4*)ldst, 0, 0);
}
__device__ __forceinline__ void async_wait0() {
  __builtin_amdgcn_s_wait_asynccnt(0);
}
#else
__device__ __forceinline__ void async_copy16(const unsigned short* gsrc,
                                             unsigned short* ldst) {
  *(uint4*)ldst = *(const uint4*)gsrc;
}
__device__ __forceinline__ void async_wait0() {}
#endif

__device__ __forceinline__ unsigned short f2bf(float f) {
  unsigned u = __float_as_uint(f);
  unsigned r = u + 0x7FFFu + ((u >> 16) & 1u);   // round-to-nearest-even
  return (unsigned short)(r >> 16);
}
__device__ __forceinline__ float bf2f(unsigned short h) {
  return __uint_as_float(((unsigned)h) << 16);
}

// A-fragment (16x32 bf16): lane m = lane&15; kb = 8*(lane>=16);
// VGPR0-3 = K[kb..kb+7], VGPR4-7 = K[kb+16..kb+23].  rowbase points at k=0.
__device__ __forceinline__ v16bf load_afrag(const unsigned short* rowbase, int lane) {
  const int kb = (lane & 16) ? 8 : 0;
  union { uint4 u[2]; v16bf v; } f;
  f.u[0] = *(const uint4*)(rowbase + kb);
  f.u[1] = *(const uint4*)(rowbase + kb + 16);
  return f.v;
}
// B-fragment (32x16 bf16), column-contiguous storage Bt[n][k]:
// lane n = lane&15; kb = 8*(lane>=16); VGPR0-3 = K[kb..kb+7], VGPR4-7 = K[kb+16..kb+23].
__device__ __forceinline__ v16bf load_bfrag(const unsigned short* colbase, int lane) {
  const int kb = (lane & 16) ? 8 : 0;
  union { uint4 u[2]; v16bf v; } f;
  f.u[0] = *(const uint4*)(colbase + kb);
  f.u[1] = *(const uint4*)(colbase + kb + 16);
  return f.v;
}
__device__ __forceinline__ v8f wmma_bf16(v16bf a, v16bf b, v8f c) {
  return __builtin_amdgcn_wmma_f32_16x16x32_bf16(false, a, false, b, (short)0, c,
                                                 false, false);
}

// ---------------------------------------------------------------------------
// fp32 -> bf16 conversion
// ---------------------------------------------------------------------------
__global__ void cvt_bf16(const float* __restrict__ in, unsigned short* __restrict__ out, int n) {
  int i = blockIdx.x * blockDim.x + threadIdx.x;
  if (i < n) out[i] = f2bf(in[i]);
}

// ---------------------------------------------------------------------------
// Generic GEMM: C[M,N] = A[M,K] * B[K,N]; A,B bf16 row-major; C f32 or bf16.
// 128 threads = 4 waves; block tile 64x128, K-step 32; wave tile 32x64.
// 8 WMMAs per wave per K-step.  A staged via async-to-LDS; B transposed
// through a b128 register batch (all loads in flight before LDS stores).
// ---------------------------------------------------------------------------
__global__ __launch_bounds__(128) void gemm_bf16(const unsigned short* __restrict__ A,
                                                 const unsigned short* __restrict__ B,
                                                 void* __restrict__ C,
                                                 int M, int N, int K, int out_f32) {
  __shared__ __align__(16) unsigned short lds_a[64 * 32];    // [m][k]   4KB
  __shared__ __align__(16) unsigned short lds_bt[128 * 32];  // [n][k]   8KB
  const int tid  = threadIdx.x;
  const int lane = tid & 31;
  const int wave = tid >> 5;
  const int m0 = blockIdx.y * 64;
  const int n0 = blockIdx.x * 128;
  const int wr = (wave >> 1) * 32;
  const int wc = (wave & 1) * 64;
  v8f acc[2][4] = {};

  for (int k0 = 0; k0 < K; k0 += 32) {
    __syncthreads();
    // --- A tile: 64x32 = 256 16B chunks, 2 per thread, async copy ---
#pragma unroll
    for (int i = 0; i < 2; i++) {
      int id = tid + 128 * i;
      int r = id >> 2, c = (id & 3) * 8;
      async_copy16(A + (size_t)(m0 + r) * K + k0 + c, lds_a + r * 32 + c);
    }
    // --- B tile: 32x128 = 512 b128 chunks, 4 per thread; transpose on store ---
    uint4 bq[4];
#pragma unroll
    for (int i = 0; i < 4; i++) {
      int id = tid + 128 * i;
      int kk = id >> 4, c = (id & 15) * 8;
      bq[i] = *(const uint4*)(B + (size_t)(k0 + kk) * N + n0 + c);
    }
#pragma unroll
    for (int i = 0; i < 4; i++) {
      int id = tid + 128 * i;
      int kk = id >> 4, c = (id & 15) * 8;
      unsigned w[4] = {bq[i].x, bq[i].y, bq[i].z, bq[i].w};
#pragma unroll
      for (int j = 0; j < 4; j++) {
        lds_bt[(c + 2 * j    ) * 32 + kk] = (unsigned short)(w[j] & 0xFFFFu);
        lds_bt[(c + 2 * j + 1) * 32 + kk] = (unsigned short)(w[j] >> 16);
      }
    }
    if (k0 + 32 < K)                          // global_prefetch_b8 of next B tile
      __builtin_prefetch(B + (size_t)(k0 + 32) * N + n0 + tid, 0, 1);
    async_wait0();
    __syncthreads();

    v16bf af[2], bfr[4];
#pragma unroll
    for (int i = 0; i < 2; i++)
      af[i] = load_afrag(lds_a + (wr + i * 16 + (lane & 15)) * 32, lane);
#pragma unroll
    for (int j = 0; j < 4; j++)
      bfr[j] = load_bfrag(lds_bt + (wc + j * 16 + (lane & 15)) * 32, lane);
#pragma unroll
    for (int i = 0; i < 2; i++)
#pragma unroll
      for (int j = 0; j < 4; j++)
        acc[i][j] = wmma_bf16(af[i], bfr[j], acc[i][j]);
  }

  const int mh = (lane & 16) ? 8 : 0;
  const int nn = lane & 15;
#pragma unroll
  for (int i = 0; i < 2; i++)
#pragma unroll
    for (int j = 0; j < 4; j++)
#pragma unroll
      for (int r = 0; r < 8; r++) {
        int row = m0 + wr + i * 16 + r + mh;
        int col = n0 + wc + j * 16 + nn;
        float v = acc[i][j][r];
        if (out_f32) ((float*)C)[(size_t)row * N + col] = v;
        else ((unsigned short*)C)[(size_t)row * N + col] = f2bf(v);
      }
}

// ---------------------------------------------------------------------------
// Interleaved RoPE, in place on bf16 Q [S][NH*HD] and K [S][NKV*HD].
// cos[d] = cos(t * theta^{-2*(d&63)/HD}); pairs (2i,2i+1) rotated.
// ---------------------------------------------------------------------------
__global__ void rope_kernel(unsigned short* __restrict__ q, unsigned short* __restrict__ k) {
  const int QP = S_LEN * NH * (HD / 2);
  const int KP = S_LEN * NKV * (HD / 2);
  int idx = blockIdx.x * blockDim.x + threadIdx.x;
  unsigned short* p;
  int s, i;
  if (idx < QP) {
    s = idx / (NH * (HD / 2));
    int r = idx % (NH * (HD / 2));
    int h = r / (HD / 2);
    i = r % (HD / 2);
    p = q + (size_t)s * (NH * HD) + h * HD + 2 * i;
  } else if (idx < QP + KP) {
    int id2 = idx - QP;
    s = id2 / (NKV * (HD / 2));
    int r = id2 % (NKV * (HD / 2));
    int h = r / (HD / 2);
    i = r % (HD / 2);
    p = k + (size_t)s * (NKV * HD) + h * HD + 2 * i;
  } else return;
  const float lnb = 9.210340371976184f;  // ln(10000)
  int d0 = 2 * i, d1 = 2 * i + 1;
  float f0 = __expf(-((float)(2 * (d0 & 63))) * (1.0f / HD) * lnb);
  float f1 = __expf(-((float)(2 * (d1 & 63))) * (1.0f / HD) * lnb);
  float t = (float)s;
  float c0 = cosf(t * f0), s0 = sinf(t * f0);
  float c1 = cosf(t * f1), s1 = sinf(t * f1);
  float x0 = bf2f(p[0]), x1 = bf2f(p[1]);
  p[0] = f2bf(x0 * c0 - x1 * s0);
  p[1] = f2bf(x1 * c1 + x0 * s1);
}

// ---------------------------------------------------------------------------
// Flash attention: block = (head h, 64 query rows), 4 waves x 16 rows.
// Key tiles of 32, online softmax, GQA via kvh = h>>2. Output bf16 [S][NH*HD].
// K tile staged with async-to-LDS; V transpose staged via b128 register batch.
// ---------------------------------------------------------------------------
__global__ __launch_bounds__(128) void flash_attn(const unsigned short* __restrict__ Q,
                                                  const unsigned short* __restrict__ Kb,
                                                  const unsigned short* __restrict__ Vb,
                                                  unsigned short* __restrict__ O) {
  __shared__ __align__(16) unsigned short lds_k[32 * 128];    // [key][d] == B-frag layout for Q*K^T
  __shared__ __align__(16) unsigned short lds_vt[128 * 32];   // [d][key] == B-frag layout for P*V
  __shared__ __align__(16) unsigned short lds_p[4][16 * 32];  // per-wave P bounce
  const int tid = threadIdx.x, lane = tid & 31, wave = tid >> 5;
  const int qb  = blockIdx.x * 64;
  const int h   = blockIdx.y;
  const int kvh = h >> 2;                 // N_REP = 4
  const int q0w = qb + wave * 16;
  const float scale = 0.08838834764831845f;  // 1/sqrt(128)

  // Q fragments for this wave's 16 rows, 4 k-chunks of 32 along d: kept in VGPRs.
  v16bf qf[4];
  {
    const unsigned short* qrow = Q + (size_t)(q0w + (lane & 15)) * (NH * HD) + h * HD;
#pragma unroll
    for (int c = 0; c < 4; c++) qf[c] = load_afrag(qrow + c * 32, lane);
  }

  v8f acc[8] = {};
  float m_i[8], l_i[8];
#pragma unroll
  for (int r = 0; r < 8; r++) { m_i[r] = -1e30f; l_i[r] = 0.f; }

  const int nkt = qb / 32 + 2;            // causal: keys <= qb+63
  for (int kt = 0; kt < nkt; kt++) {
    const int k0 = kt * 32;
    __syncthreads();
    // --- K tile: 32x128 = 512 16B chunks, 4 per thread, async copy ---
#pragma unroll
    for (int i = 0; i < 4; i++) {
      int id = tid + 128 * i;
      int r = id >> 4, c = (id & 15) * 8;
      async_copy16(Kb + (size_t)(k0 + r) * (NKV * HD) + kvh * HD + c,
                   lds_k + r * 128 + c);
    }
    // --- V tile transposed: 512 b128 chunks, 4 per thread; transpose on store ---
    uint4 vq[4];
#pragma unroll
    for (int i = 0; i < 4; i++) {
      int id = tid + 128 * i;
      int r = id >> 4, c = (id & 15) * 8;
      vq[i] = *(const uint4*)(Vb + (size_t)(k0 + r) * (NKV * HD) + kvh * HD + c);
    }
#pragma unroll
    for (int i = 0; i < 4; i++) {
      int id = tid + 128 * i;
      int r = id >> 4, c = (id & 15) * 8;
      unsigned w[4] = {vq[i].x, vq[i].y, vq[i].z, vq[i].w};
#pragma unroll
      for (int j = 0; j < 4; j++) {
        lds_vt[(c + 2 * j    ) * 32 + r] = (unsigned short)(w[j] & 0xFFFFu);
        lds_vt[(c + 2 * j + 1) * 32 + r] = (unsigned short)(w[j] >> 16);
      }
    }
    async_wait0();
    __syncthreads();

    // S = Q * K^T : two 16x16 score tiles (keys k0..+15, k0+16..+31)
    v8f sc0 = {}, sc1 = {};
#pragma unroll
    for (int c = 0; c < 4; c++) {
      v16bf b0 = load_bfrag(lds_k + ((lane & 15)     ) * 128 + c * 32, lane);
      v16bf b1 = load_bfrag(lds_k + ((lane & 15) + 16) * 128 + c * 32, lane);
      sc0 = wmma_bf16(qf[c], b0, sc0);
      sc1 = wmma_bf16(qf[c], b1, sc1);
    }

    // Online softmax per row (row m = r + 8*(lane>=16), spread over a 16-lane half)
    unsigned short* pbuf = lds_p[wave];
    float alpha_[8];
#pragma unroll
    for (int r = 0; r < 8; r++) {
      const int m   = r + ((lane & 16) ? 8 : 0);
      const int qg  = q0w + m;
      const int key0 = k0 + (lane & 15);
      const int key1 = key0 + 16;
      float v0 = sc0[r] * scale + ((key0 > qg) ? -1e9f : 0.f);
      float v1 = sc1[r] * scale + ((key1 > qg) ? -1e9f : 0.f);
      float mt = fmaxf(v0, v1);
#pragma unroll
      for (int off = 1; off < 16; off <<= 1) mt = fmaxf(mt, __shfl_xor(mt, off, 32));
      float mnew = fmaxf(m_i[r], mt);
      float al = __expf(m_i[r] - mnew);
      float p0 = __expf(v0 - mnew);
      float p1 = __expf(v1 - mnew);
      float rs = p0 + p1;
#pragma unroll
      for (int off = 1; off < 16; off <<= 1) rs += __shfl_xor(rs, off, 32);
      l_i[r] = l_i[r] * al + rs;
      m_i[r] = mnew;
      alpha_[r] = al;
      pbuf[m * 32 +      (lane & 15)] = f2bf(p0);
      pbuf[m * 32 + 16 + (lane & 15)] = f2bf(p1);
    }
#pragma unroll
    for (int nt = 0; nt < 8; nt++)
#pragma unroll
      for (int r = 0; r < 8; r++) acc[nt][r] *= alpha_[r];
    __syncthreads();   // make pbuf (cross-lane) visible

    // O += P(16x32) * V(32x128): P via LDS bounce -> A-frag; V^T -> B-frags
    v16bf pa = load_afrag(pbuf + (lane & 15) * 32, lane);
#pragma unroll
    for (int nt = 0; nt < 8; nt++) {
      v16bf bv = load_bfrag(lds_vt + (nt * 16 + (lane & 15)) * 32, lane);
      acc[nt] = wmma_bf16(pa, bv, acc[nt]);
    }
  }

  // epilogue: divide by l, store bf16
  const int mh = (lane & 16) ? 8 : 0, nn = lane & 15;
#pragma unroll
  for (int nt = 0; nt < 8; nt++)
#pragma unroll
    for (int r = 0; r < 8; r++) {
      int row = q0w + r + mh;
      int col = h * HD + nt * 16 + nn;
      O[(size_t)row * (NH * HD) + col] = f2bf(acc[nt][r] / l_i[r]);
    }
}

// ---------------------------------------------------------------------------
extern "C" void kernel_launch(void* const* d_in, const int* in_sizes, int n_in,
                              void* d_out, int out_size, void* d_ws, size_t ws_size,
                              hipStream_t stream) {
  (void)in_sizes; (void)n_in; (void)out_size; (void)ws_size;
  const float* hs = (const float*)d_in[0];
  // d_in[1] = attention_mask: causal applied analytically, not read.
  const float* wq = (const float*)d_in[2];
  const float* wk = (const float*)d_in[3];
  const float* wv = (const float*)d_in[4];
  const float* wo = (const float*)d_in[5];

  unsigned short* base = (unsigned short*)d_ws;
  size_t off = 0;
  unsigned short* Hb  = base + off; off += (size_t)S_LEN * HIDN;        // hidden bf16
  unsigned short* Wqb = base + off; off += (size_t)HIDN * NH * HD;
  unsigned short* Wkb = base + off; off += (size_t)HIDN * NKV * HD;
  unsigned short* Wvb = base + off; off += (size_t)HIDN * NKV * HD;
  unsigned short* Wob = base + off; off += (size_t)NH * HD * HIDN;
  unsigned short* Ql  = base + off; off += (size_t)S_LEN * NH * HD;
  unsigned short* Kl  = base + off; off += (size_t)S_LEN * NKV * HD;
  unsigned short* Vl  = base + off; off += (size_t)S_LEN * NKV * HD;
  unsigned short* Ao  = base + off; off += (size_t)S_LEN * NH * HD;

  auto cvt = [&](const float* src, unsigned short* dst, size_t n) {
    cvt_bf16<<<dim3((unsigned)((n + 255) / 256)), 256, 0, stream>>>(src, dst, (int)n);
  };
  cvt(hs, Hb, (size_t)S_LEN * HIDN);
  cvt(wq, Wqb, (size_t)HIDN * NH * HD);
  cvt(wk, Wkb, (size_t)HIDN * NKV * HD);
  cvt(wv, Wvb, (size_t)HIDN * NKV * HD);
  cvt(wo, Wob, (size_t)NH * HD * HIDN);

  gemm_bf16<<<dim3((NH * HD) / 128, S_LEN / 64), 128, 0, stream>>>(Hb, Wqb, Ql, S_LEN, NH * HD, HIDN, 0);
  gemm_bf16<<<dim3((NKV * HD) / 128, S_LEN / 64), 128, 0, stream>>>(Hb, Wkb, Kl, S_LEN, NKV * HD, HIDN, 0);
  gemm_bf16<<<dim3((NKV * HD) / 128, S_LEN / 64), 128, 0, stream>>>(Hb, Wvb, Vl, S_LEN, NKV * HD, HIDN, 0);

  {
    int total = S_LEN * NH * (HD / 2) + S_LEN * NKV * (HD / 2);
    rope_kernel<<<(total + 255) / 256, 256, 0, stream>>>(Ql, Kl);
  }

  flash_attn<<<dim3(S_LEN / 64, NH), 128, 0, stream>>>(Ql, Kl, Vl, Ao);

  gemm_bf16<<<dim3(HIDN / 128, S_LEN / 64), 128, 0, stream>>>(Ao, Wob, d_out, S_LEN, HIDN, NH * HD, 1);
}
